// VectorQuantizer_5342939316377
// MI455X (gfx1250) — matrix operand order repaired and checked by reading, not compiled, not run on hardware
//
#include <hip/hip_runtime.h>

// ---------------------------------------------------------------------------
// VQ-VAE vector quantizer for MI455X (gfx1250, wave32, WMMA).
//   inputs : x [32768 x 256] f32, codebook [4096 x 256] f32
//   outputs: z_q_st [32768*256] f32, loss [1] f32, indices [32768] (as f32)
// Strategy: bf16x3 split GEMM (v_wmma_f32_16x16x32_bf16), fused argmin,
//           async-to-LDS double-buffered codebook tiles (ASYNCcnt path).
// ---------------------------------------------------------------------------

typedef __attribute__((ext_vector_type(16))) __bf16 v16bf;
typedef __attribute__((ext_vector_type(8)))  __bf16 v8bf;
typedef __attribute__((ext_vector_type(8)))  float  v8f;
typedef __attribute__((ext_vector_type(4)))  int    v4i;

typedef __attribute__((address_space(1))) v4i g_v4i;   // global int32x4
typedef __attribute__((address_space(3))) v4i l_v4i;   // LDS int32x4

#define NUM_EMB 4096
#define DDIM    256
#define N_ROWS  32768
#define BM      128                // rows per workgroup (8 row-tiles x 16)
#define NWG     (N_ROWS / BM)      // 256 workgroups
#define BN      16                 // codebook columns per K-step
#define KTILES  (NUM_EMB / BN)     // 256 tiles
#define LDS_STR 264                // padded bf16 stride (bank-conflict dodge)

#if __has_builtin(__builtin_amdgcn_global_load_async_to_lds_b128)
#define HAVE_ASYNC 1
#else
#define HAVE_ASYNC 0
#endif

static __device__ __forceinline__ void await_async() {
#if HAVE_ASYNC
#if __has_builtin(__builtin_amdgcn_s_wait_asynccnt)
    __builtin_amdgcn_s_wait_asynccnt(0);
#else
    asm volatile("s_wait_asynccnt 0x0" ::: "memory");
#endif
#endif
}

static __device__ __forceinline__ unsigned int fkey(float f) {
    // monotonic float -> uint map (ascending), so u64 min == (val,idx) argmin
    unsigned int u = __float_as_uint(f);
    return (u & 0x80000000u) ? ~u : (u | 0x80000000u);
}

// Stage one 16-column codebook tile (hi rows 0..15, lo rows 16..31) into LDS.
// 256 threads x 2 iters x (hi+lo) b128 = 16KB per tile.
static __device__ __forceinline__ void stage_tile(
    int tid, int jbase,
    const __bf16* __restrict__ cb_hi, const __bf16* __restrict__ cb_lo,
    __bf16* dst)
{
    #pragma unroll
    for (int i = 0; i < 2; ++i) {
        const int q   = tid + i * 256;          // 0..511 : 16-byte chunk id
        const int col = q >> 5;                 // 0..15
        const int off = (q & 31) * 8;           // 0..248 bf16 elems
        const size_t gofs = (size_t)(jbase + col) * DDIM + off;
        __bf16* lh = dst + col * LDS_STR + off;
        __bf16* ll = dst + (16 + col) * LDS_STR + off;
#if HAVE_ASYNC
        void* gh = (void*)(cb_hi + gofs);       // drop const, then cast AS/type
        void* gl = (void*)(cb_lo + gofs);
        __builtin_amdgcn_global_load_async_to_lds_b128(
            (g_v4i*)gh, (l_v4i*)lh, 0, 0);
        __builtin_amdgcn_global_load_async_to_lds_b128(
            (g_v4i*)gl, (l_v4i*)ll, 0, 0);
#else
        *(float4*)lh = *(const float4*)(cb_hi + gofs);
        *(float4*)ll = *(const float4*)(cb_lo + gofs);
#endif
    }
}

// ---------------------------------------------------------------------------
// Kernel 1: split codebook into bf16 hi/lo and compute ||e||^2 per code.
// ---------------------------------------------------------------------------
__global__ __launch_bounds__(256)
void vq_prep(const float* __restrict__ cb, __bf16* __restrict__ cb_hi,
             __bf16* __restrict__ cb_lo, float* __restrict__ e_sq)
{
    __shared__ float sr[256];
    const int j = blockIdx.x;
    const int t = threadIdx.x;
    const float v  = cb[(size_t)j * DDIM + t];
    const __bf16 h = (__bf16)v;
    cb_hi[(size_t)j * DDIM + t] = h;
    cb_lo[(size_t)j * DDIM + t] = (__bf16)(v - (float)h);
    sr[t] = v * v;
    __syncthreads();
    #pragma unroll
    for (int s = 128; s > 0; s >>= 1) {
        if (t < s) sr[t] += sr[t + s];
        __syncthreads();
    }
    if (t == 0) e_sq[j] = sr[0];
}

// ---------------------------------------------------------------------------
// Kernel 2: GEMM + argmin + gather + partial loss.
// grid 256 x 256 threads. 8 waves = 8 row-tiles sharing one B tile from LDS.
// ---------------------------------------------------------------------------
__global__ __launch_bounds__(256)
void vq_main(const float* __restrict__ x, const float* __restrict__ cb,
             const __bf16* __restrict__ cb_hi, const __bf16* __restrict__ cb_lo,
             const float* __restrict__ e_sq, float* __restrict__ out,
             float* __restrict__ partials)
{
    __shared__ __align__(16) __bf16 sB[2][32][LDS_STR];   // [buf][hi16|lo16][k]
    __shared__ unsigned long long sBest[BM];
    __shared__ float sRed[256];

    const int tid  = threadIdx.x;
    const int lane = tid & 31;
    const int r    = tid >> 5;                // row tile 0..7
    const int rowBase = blockIdx.x * BM;
    const int laneRow = rowBase + r * 16 + (lane & 15);
    const int aKoff = (lane & 16) ? 8  : 0;   // A: lanes>=16 own K {8..15,24..31}
    const int bKoff = (lane & 16) ? 8  : 0;   // B: lanes>=16 own K {8..15,24..31}
    const int bcol  = lane & 15;

    if (tid < BM) sBest[tid] = ~0ull;

    // ---- load x row fragments once, split into bf16 hi/lo, keep in VGPRs ----
    v16bf a_hi[8], a_lo[8];
    #pragma unroll
    for (int d = 0; d < 8; ++d) {
        const float* ap = x + (size_t)laneRow * DDIM + d * 32 + aKoff;
        #pragma unroll
        for (int h = 0; h < 2; ++h) {         // h=0: K+0..7, h=1: K+16..23
            const float4 f0 = *(const float4*)(ap + h * 16);
            const float4 f1 = *(const float4*)(ap + h * 16 + 4);
            const float tmp[8] = {f0.x, f0.y, f0.z, f0.w, f1.x, f1.y, f1.z, f1.w};
            #pragma unroll
            for (int i = 0; i < 8; ++i) {
                const __bf16 hi = (__bf16)tmp[i];
                a_hi[d][h * 8 + i] = hi;
                a_lo[d][h * 8 + i] = (__bf16)(tmp[i] - (float)hi);
            }
        }
    }

    float best_val[8];
    int   best_idx[8];
    #pragma unroll
    for (int v = 0; v < 8; ++v) { best_val[v] = 3.4e38f; best_idx[v] = 0; }

    // ---- prologue: DMA tile 0, then pipelined sweep over 4096 codes ----
    stage_tile(tid, 0, cb_hi, cb_lo, &sB[0][0][0]);

    for (int jt = 0; jt < KTILES; ++jt) {
        await_async();                         // tile jt resident in LDS
        __syncthreads();                       // ...visible to all waves
        if (jt + 1 < KTILES)                   // overlap next DMA with compute
            stage_tile(tid, (jt + 1) * BN, cb_hi, cb_lo, &sB[(jt + 1) & 1][0][0]);

        const __bf16* bh = &sB[jt & 1][bcol][bKoff];
        const __bf16* bl = &sB[jt & 1][16 + bcol][bKoff];

        v8f accHH = {}, accLH = {}, accHL = {};          // 3 independent chains
        #pragma unroll
        for (int d = 0; d < 8; ++d) {
            union { v16bf v; v8bf h[2]; } bhv, blv;      // 16B LDS loads
            bhv.h[0] = *(const v8bf*)(bh + d * 32);
            bhv.h[1] = *(const v8bf*)(bh + d * 32 + 16);
            blv.h[0] = *(const v8bf*)(bl + d * 32);
            blv.h[1] = *(const v8bf*)(bl + d * 32 + 16);
            accHH = __builtin_amdgcn_wmma_f32_16x16x32_bf16(
                        false, a_hi[d], false, bhv.v, (short)0, accHH, false, false);
            accLH = __builtin_amdgcn_wmma_f32_16x16x32_bf16(
                        false, a_lo[d], false, bhv.v, (short)0, accLH, false, false);
            accHL = __builtin_amdgcn_wmma_f32_16x16x32_bf16(
                        false, a_hi[d], false, blv.v, (short)0, accHL, false, false);
        }
        const int j = jt * BN + bcol;
        const float ej = e_sq[j];
        #pragma unroll
        for (int v = 0; v < 8; ++v) {
            // distance (minus row-constant ||z||^2): ||e||^2 - 2 z.e
            const float dot = accHH[v] + accLH[v] + accHL[v];
            const float val = fmaf(-2.0f, dot, ej);
            if (val < best_val[v]) { best_val[v] = val; best_idx[v] = j; }
        }
    }

    __syncthreads();
    #pragma unroll
    for (int v = 0; v < 8; ++v) {
        const int lrow = r * 16 + v + ((lane & 16) ? 8 : 0);   // C layout rows
        const unsigned long long pk =
            ((unsigned long long)fkey(best_val[v]) << 32) |
            (unsigned long long)(unsigned int)best_idx[v];
        atomicMin(&sBest[lrow], pk);
    }
    __syncthreads();

    // ---- outputs: straight-through z_q, indices, loss partial ----
    float lsum = 0.f;
    for (int e = tid; e < BM * DDIM; e += 256) {
        const int lrow = e >> 8;
        const int k    = e & (DDIM - 1);
        const int idx  = (int)(unsigned int)(sBest[lrow] & 0xffffffffull);
        const size_t n = (size_t)(rowBase + lrow);
        const float zq = cb[(size_t)idx * DDIM + k];
        const float xv = x[n * DDIM + k];
        out[n * DDIM + k] = xv + (zq - xv);    // matches x + sg(zq - x)
        const float diff = zq - xv;
        lsum += diff * diff;
    }
    if (tid < BM) {
        const int idx = (int)(unsigned int)(sBest[tid] & 0xffffffffull);
        out[(size_t)N_ROWS * DDIM + 1 + rowBase + tid] = (float)idx;
    }
    sRed[tid] = lsum;
    __syncthreads();
    #pragma unroll
    for (int s = 128; s > 0; s >>= 1) {
        if (tid < s) sRed[tid] += sRed[tid + s];
        __syncthreads();
    }
    if (tid == 0) partials[blockIdx.x] = sRed[0];
}

// ---------------------------------------------------------------------------
// Kernel 3: deterministic fixed-order loss reduction across 256 partials.
// loss = 1.25 * sum((zq-x)^2) / (N*D)
// ---------------------------------------------------------------------------
__global__ __launch_bounds__(256)
void vq_loss(const float* __restrict__ partials, float* __restrict__ out)
{
    __shared__ float sr[256];
    const int t = threadIdx.x;
    sr[t] = partials[t];
    __syncthreads();
    #pragma unroll
    for (int s = 128; s > 0; s >>= 1) {
        if (t < s) sr[t] += sr[t + s];
        __syncthreads();
    }
    if (t == 0)
        out[(size_t)N_ROWS * DDIM] =
            1.25f * sr[0] * (1.0f / ((float)N_ROWS * (float)DDIM));
}

// ---------------------------------------------------------------------------
extern "C" void kernel_launch(void* const* d_in, const int* in_sizes, int n_in,
                              void* d_out, int out_size, void* d_ws, size_t ws_size,
                              hipStream_t stream)
{
    const float* x  = (const float*)d_in[0];
    const float* cb = (const float*)d_in[1];

    // workspace: cb_hi (2MB) | cb_lo (2MB) | e_sq (16KB) | partials (1KB)
    __bf16* cb_hi    = (__bf16*)d_ws;
    __bf16* cb_lo    = cb_hi + (size_t)NUM_EMB * DDIM;
    float*  e_sq     = (float*)(cb_lo + (size_t)NUM_EMB * DDIM);
    float*  partials = e_sq + NUM_EMB;
    float*  out      = (float*)d_out;

    vq_prep<<<NUM_EMB, 256, 0, stream>>>(cb, cb_hi, cb_lo, e_sq);
    vq_main<<<NWG, 256, 0, stream>>>(x, cb, cb_hi, cb_lo, e_sq, out, partials);
    vq_loss<<<1, 256, 0, stream>>>(partials, out);
}
